// AtnConv_1039382086318
// MI455X (gfx1250) — compile-verified, matrix operand order checked by reading, not run on hardware
//
#include <hip/hip_runtime.h>
#include <hip/hip_bf16.h>

// ---------------------------------------------------------------------------
// Contextual attention (AtnConv) for MI455X / gfx1250.
// Dominant cost: two 4096x4096x1152-class GEMMs per batch -> bf16 WMMA
// (v_wmma_f32_16x16x32_bf16, f32 accumulate), with async global->LDS DMA
// (GLOBAL_LOAD_ASYNC_TO_LDS_B128 / ASYNCcnt) double-buffering the K panels.
// Everything else is cheap element-wise / reduction work kept in f32.
// Workspace requirement: ~147 MB (see layout in kernel_launch).
// ---------------------------------------------------------------------------

typedef __attribute__((ext_vector_type(16))) __bf16        v16bf;
typedef __attribute__((ext_vector_type(8)))  float         v8f;
typedef __attribute__((ext_vector_type(4)))  unsigned int  u32x4;

#define L2PATCH 4096   // 64*64 patch positions
#define K9C     1152   // 3*3*128 patch length
#define CCH     128
#define H2      64
#define H1      128
#define LDSK    40     // padded LDS row stride (bf16 elems) for 32-wide K tiles

union Frag   { u32x4 u[2]; v16bf v; };
union Pack16 { u32x4 u[2]; unsigned short s[16]; };

static __device__ __forceinline__ unsigned short f2bf(float f) {
  unsigned int u = __float_as_uint(f);
  u += 0x7fffu + ((u >> 16) & 1u);          // round-to-nearest-even
  return (unsigned short)(u >> 16);
}

// ---------------------------------------------------------------------------
// gfx1250 async DMA: copy 16 bytes per lane from global to LDS, tracked by
// ASYNCcnt. The LDS operand is the low 32 bits of the generic pointer
// (LDS aperture keeps the byte offset in addr[31:0]).
// ---------------------------------------------------------------------------
static __device__ __forceinline__ void async_copy_b128(void* lds_dst,
                                                       const void* gsrc) {
  const unsigned lds = (unsigned)(unsigned long long)lds_dst;
  const unsigned long long ga = (unsigned long long)gsrc;
  asm volatile("global_load_async_to_lds_b128 %0, %1, off"
               :: "v"(lds), "v"(ga)
               : "memory");
}

static __device__ __forceinline__ void wait_async_all() {
  asm volatile("s_wait_asynccnt 0" ::: "memory");
}

// ---------------------------------------------------------------------------
// im2col of x2 (stride 1, SAME, pt=1) -> bf16 rows + 1/max(||row||,1e-4)
// one block (128 threads) per patch position
// ---------------------------------------------------------------------------
__global__ __launch_bounds__(128) void k_im2col_x2(
    const float* __restrict__ x2, int b,
    unsigned short* __restrict__ A2, float* __restrict__ inv_norm)
{
  __shared__ float red[128];
  const int l = blockIdx.x;
  const int i = l >> 6, j = l & 63;
  const int c = threadIdx.x;
  float ssq = 0.f;
#pragma unroll
  for (int tap = 0; tap < 9; ++tap) {
    const int di = tap / 3, dj = tap % 3;
    const int ii = i + di - 1, jj = j + dj - 1;
    float v = 0.f;
    if (ii >= 0 && ii < H2 && jj >= 0 && jj < H2)
      v = x2[(((size_t)b * H2 + ii) * H2 + jj) * CCH + c];
    A2[(size_t)l * K9C + tap * CCH + c] = f2bf(v);
    ssq += v * v;
  }
  red[c] = ssq;
  __syncthreads();
  for (int s = 64; s > 0; s >>= 1) {
    if (c < s) red[c] += red[c + s];
    __syncthreads();
  }
  if (c == 0) inv_norm[l] = 1.f / fmaxf(sqrtf(red[0]), 1e-4f);
}

// ---------------------------------------------------------------------------
// im2col of x1 (stride 2, SAME, pt=0) -> bf16 rows ("raw" patches)
// ---------------------------------------------------------------------------
__global__ __launch_bounds__(128) void k_im2col_x1(
    const float* __restrict__ x1, int b, unsigned short* __restrict__ A1)
{
  const int l = blockIdx.x;
  const int i = l >> 6, j = l & 63;
  const int c = threadIdx.x;
#pragma unroll
  for (int tap = 0; tap < 9; ++tap) {
    const int di = tap / 3, dj = tap % 3;
    const int ii = 2 * i + di, jj = 2 * j + dj;
    float v = 0.f;
    if (ii < H1 && jj < H1)
      v = x1[(((size_t)b * H1 + ii) * H1 + jj) * CCH + c];
    A1[(size_t)l * K9C + tap * CCH + c] = f2bf(v);
  }
}

// ---------------------------------------------------------------------------
// mi[l] = (mean of 3x3 SAME patch of mask == 1.0)  (bilinear 64->64 = identity)
// ---------------------------------------------------------------------------
__global__ void k_mask_mi(const float* __restrict__ mask, int b,
                          float* __restrict__ mi)
{
  const int l = blockIdx.x * blockDim.x + threadIdx.x;
  if (l >= L2PATCH) return;
  const int i = l >> 6, j = l & 63;
  float s = 0.f;
#pragma unroll
  for (int di = 0; di < 3; ++di)
#pragma unroll
    for (int dj = 0; dj < 3; ++dj) {
      const int ii = i + di - 1, jj = j + dj - 1;
      if (ii >= 0 && ii < H2 && jj >= 0 && jj < H2)
        s += mask[((size_t)b * H2 + ii) * H2 + jj];
    }
  mi[l] = ((s / 9.0f) == 1.0f) ? 1.f : 0.f;
}

// ---------------------------------------------------------------------------
// WMMA fragment loader. LDS tile is [128 rows][LDSK] bf16, rows are M (or N),
// columns are K. 16-bit operand lane layout (ISA 7.12.2):
//   lane<16 : row = lane,    K = 0..7 and 16..23
//   lane>=16: row = lane-16, K = 8..15 and 24..31
// ---------------------------------------------------------------------------
static __device__ __forceinline__ v16bf load_frag(
    const unsigned short* __restrict__ tile, int row0, int lane)
{
  const int fm = lane & 15;
  const int fk = (lane >> 4) * 8;
  const int r  = (row0 + fm) * LDSK;
  Frag f;
  f.u[0] = *reinterpret_cast<const u32x4*>(&tile[r + fk]);
  f.u[1] = *reinterpret_cast<const u32x4*>(&tile[r + fk + 16]);
  return f.v;
}

// ---------------------------------------------------------------------------
// GEMM1: S = A2 * A2^T   (M=N=4096, K=1152), bf16 in / f32 out.
// 128x128 tile per block, 8 waves in 2(M) x 4(N), each wave 64x32
// = 4x2 wmma tiles. BK = 32. Both K panels are staged with async DMA,
// double-buffered so tile k+1 streams in while tile k feeds the WMMAs.
// ---------------------------------------------------------------------------
__global__ __launch_bounds__(256) void k_gemm_scores(
    const unsigned short* __restrict__ A2, float* __restrict__ S)
{
  __shared__ unsigned short At[2][128 * LDSK];
  __shared__ unsigned short Bt[2][128 * LDSK];
  const int tid  = threadIdx.x;
  const int lane = tid & 31;
  const int wave = tid >> 5;
  const int wm = wave >> 2;                 // 0..1
  const int wn = wave & 3;                  // 0..3
  const int m0 = blockIdx.x * 128;
  const int n0 = blockIdx.y * 128;
  const int srow = tid >> 1;                // staging: 0..127
  const int scol = (tid & 1) * 16;          // staging: 0 or 16

  const unsigned short* gA = &A2[(size_t)(m0 + srow) * K9C + scol];
  const unsigned short* gB = &A2[(size_t)(n0 + srow) * K9C + scol];
  const int lo = srow * LDSK + scol;        // LDS element offset

  const v8f vz = {0.f, 0.f, 0.f, 0.f, 0.f, 0.f, 0.f, 0.f};
  v8f acc[4][2];
#pragma unroll
  for (int i = 0; i < 4; ++i)
#pragma unroll
    for (int j = 0; j < 2; ++j) acc[i][j] = vz;

  // prologue: stream tile kk=0 into buffer 0
  async_copy_b128(&At[0][lo],     gA);
  async_copy_b128(&At[0][lo + 8], gA + 8);
  async_copy_b128(&Bt[0][lo],     gB);
  async_copy_b128(&Bt[0][lo + 8], gB + 8);

  int buf = 0;
  for (int kk = 0; kk < K9C; kk += 32) {
    wait_async_all();          // my DMA for tile kk landed
    __syncthreads();           // everyone's landed; prior-buffer reads retired
    if (kk + 32 < K9C) {       // stream tile kk+32 while computing tile kk
      async_copy_b128(&At[buf ^ 1][lo],     gA + kk + 32);
      async_copy_b128(&At[buf ^ 1][lo + 8], gA + kk + 40);
      async_copy_b128(&Bt[buf ^ 1][lo],     gB + kk + 32);
      async_copy_b128(&Bt[buf ^ 1][lo + 8], gB + kk + 40);
    }

    v16bf afrag[4], bfrag[2];
#pragma unroll
    for (int i = 0; i < 4; ++i)
      afrag[i] = load_frag(At[buf], wm * 64 + i * 16, lane);
#pragma unroll
    for (int j = 0; j < 2; ++j)
      bfrag[j] = load_frag(Bt[buf], wn * 32 + j * 16, lane);
#pragma unroll
    for (int i = 0; i < 4; ++i)
#pragma unroll
      for (int j = 0; j < 2; ++j)
        acc[i][j] = __builtin_amdgcn_wmma_f32_16x16x32_bf16(
            false, afrag[i], false, bfrag[j], (short)0, acc[i][j], false, false);
    buf ^= 1;
  }

  // D layout: VGPR v -> M = v + 8*(lane>=16), N = lane&15
#pragma unroll
  for (int i = 0; i < 4; ++i) {
    const int mBase = m0 + wm * 64 + i * 16 + ((lane >> 4) << 3);
#pragma unroll
    for (int j = 0; j < 2; ++j) {
      const int n = n0 + wn * 32 + j * 16 + (lane & 15);
#pragma unroll
      for (int v = 0; v < 8; ++v)
        S[(size_t)(mBase + v) * L2PATCH + n] = acc[i][j][v];
    }
  }
}

// ---------------------------------------------------------------------------
// Row softmax over patch axis with column norm + mask, scale 10, clip.
// One block (256 threads) per row; output bf16 for GEMM2.
// ---------------------------------------------------------------------------
__global__ __launch_bounds__(256) void k_softmax(
    const float* __restrict__ S, const float* __restrict__ inv_norm,
    const float* __restrict__ mi, unsigned short* __restrict__ Sb)
{
  __shared__ float red[256];
  const int p = blockIdx.x;
  const int tid = threadIdx.x;
  float v[16];
  float mx = -3.0e38f;
#pragma unroll
  for (int t = 0; t < 16; ++t) {
    const int l = t * 256 + tid;
    const float val = S[(size_t)p * L2PATCH + l] * inv_norm[l] * mi[l];
    v[t] = val;
    mx = fmaxf(mx, val);
  }
  red[tid] = mx; __syncthreads();
  for (int s = 128; s > 0; s >>= 1) {
    if (tid < s) red[tid] = fmaxf(red[tid], red[tid + s]);
    __syncthreads();
  }
  const float gmax = red[0];
  __syncthreads();
  float sum = 0.f;
#pragma unroll
  for (int t = 0; t < 16; ++t) {
    const float e = __expf(10.0f * (v[t] - gmax));
    v[t] = e;
    sum += e;
  }
  red[tid] = sum; __syncthreads();
  for (int s = 128; s > 0; s >>= 1) {
    if (tid < s) red[tid] += red[tid + s];
    __syncthreads();
  }
  const float inv = 1.0f / red[0];
#pragma unroll
  for (int t = 0; t < 16; ++t) {
    const int l = t * 256 + tid;
    float o = mi[l] * v[t] * inv;
    o = fminf(fmaxf(o, 1e-7f), 1e9f);
    Sb[(size_t)p * L2PATCH + l] = f2bf(o);
  }
}

// ---------------------------------------------------------------------------
// GEMM2: col = Sb * A1   (M=4096, N=1152, K=4096), bf16 in / f32 out.
// A panel (attention rows, the big stream) async + double-buffered; B panel
// is staged transposed into LDS through registers (DMA cannot transpose).
// ---------------------------------------------------------------------------
__global__ __launch_bounds__(256) void k_gemm_apply(
    const unsigned short* __restrict__ Sb,   // [4096][4096]
    const unsigned short* __restrict__ A1,   // [4096][1152]
    float* __restrict__ col)                 // [4096][1152]
{
  __shared__ unsigned short At[2][128 * LDSK];
  __shared__ unsigned short Bt[128 * LDSK];
  const int tid  = threadIdx.x;
  const int lane = tid & 31;
  const int wave = tid >> 5;
  const int wm = wave >> 2;
  const int wn = wave & 3;
  const int m0 = blockIdx.x * 128;          // rows of Sb
  const int n0 = blockIdx.y * 128;          // cols of A1 (9 tiles)
  const int srow = tid >> 1;
  const int scol = (tid & 1) * 16;
  const int tk = tid >> 3;                  // 0..31 : k row of B panel
  const int tn = (tid & 7) * 16;            // 0..112: n chunk of B panel

  const unsigned short* gA = &Sb[(size_t)(m0 + srow) * L2PATCH + scol];
  const int lo = srow * LDSK + scol;

  const v8f vz = {0.f, 0.f, 0.f, 0.f, 0.f, 0.f, 0.f, 0.f};
  v8f acc[4][2];
#pragma unroll
  for (int i = 0; i < 4; ++i)
#pragma unroll
    for (int j = 0; j < 2; ++j) acc[i][j] = vz;

  // prologue: stream A tile kk=0 into buffer 0
  async_copy_b128(&At[0][lo],     gA);
  async_copy_b128(&At[0][lo + 8], gA + 8);

  int buf = 0;
  for (int kk = 0; kk < L2PATCH; kk += 32) {
    // B panel global loads issued before the async wait so they overlap it
    Pack16 pb;
    const u32x4* gb =
        reinterpret_cast<const u32x4*>(&A1[(size_t)(kk + tk) * K9C + n0 + tn]);
    pb.u[0] = gb[0]; pb.u[1] = gb[1];
    if (kk + 32 < L2PATCH)   // gfx1250 global_prefetch_b8 on next B panel
      __builtin_prefetch(&A1[(size_t)(kk + 32 + tk) * K9C + n0 + tn], 0, 1);

    wait_async_all();
    __syncthreads();         // At[buf] landed everywhere; old Bt reads retired
    if (kk + 32 < L2PATCH) { // stream next A tile while we compute this one
      async_copy_b128(&At[buf ^ 1][lo],     gA + kk + 32);
      async_copy_b128(&At[buf ^ 1][lo + 8], gA + kk + 40);
    }
#pragma unroll
    for (int e = 0; e < 16; ++e)            // transpose into [n][k]
      Bt[(tn + e) * LDSK + tk] = pb.s[e];
    __syncthreads();         // Bt ready (At[buf] untouched by the DMA)

    v16bf afrag[4], bfrag[2];
#pragma unroll
    for (int i = 0; i < 4; ++i)
      afrag[i] = load_frag(At[buf], wm * 64 + i * 16, lane);
#pragma unroll
    for (int j = 0; j < 2; ++j)
      bfrag[j] = load_frag(Bt, wn * 32 + j * 16, lane);
#pragma unroll
    for (int i = 0; i < 4; ++i)
#pragma unroll
      for (int j = 0; j < 2; ++j)
        acc[i][j] = __builtin_amdgcn_wmma_f32_16x16x32_bf16(
            false, afrag[i], false, bfrag[j], (short)0, acc[i][j], false, false);
    buf ^= 1;
  }

#pragma unroll
  for (int i = 0; i < 4; ++i) {
    const int mBase = m0 + wm * 64 + i * 16 + ((lane >> 4) << 3);
#pragma unroll
    for (int j = 0; j < 2; ++j) {
      const int n = n0 + wn * 32 + j * 16 + (lane & 15);
#pragma unroll
      for (int v = 0; v < 8; ++v)
        col[(size_t)(mBase + v) * K9C + n] = acc[i][j][v];
    }
  }
}

// ---------------------------------------------------------------------------
// col2im: VJP of stride-2 SAME 3x3 conv (pt=0) expressed as gather; /4.
// One thread per (i1,j1,c) of one batch image.
// ---------------------------------------------------------------------------
__global__ void k_col2im(const float* __restrict__ col, float* __restrict__ y)
{
  const int idx = blockIdx.x * 256 + threadIdx.x;
  if (idx >= H1 * H1 * CCH) return;
  const int c  = idx & 127;
  const int j1 = (idx >> 7) & 127;
  const int i1 = idx >> 14;
  float s = 0.f;
#pragma unroll
  for (int di = 0; di < 3; ++di) {
    const int ti = i1 - di;
    if (ti < 0 || (ti & 1)) continue;
    const int i2 = ti >> 1;
    if (i2 >= 64) continue;
#pragma unroll
    for (int dj = 0; dj < 3; ++dj) {
      const int tj = j1 - dj;
      if (tj < 0 || (tj & 1)) continue;
      const int j2 = tj >> 1;
      if (j2 >= 64) continue;
      s += col[(size_t)(i2 * 64 + j2) * K9C + (di * 3 + dj) * CCH + c];
    }
  }
  y[idx] = s * 0.25f;
}

// ---------------------------------------------------------------------------
// Fusion: four dilated 3x3x128->16 convs (rates 1,2,4,8) + bias + ReLU,
// concatenated on channels. One block per pixel; 64 threads = (rate, oc).
// ---------------------------------------------------------------------------
__global__ __launch_bounds__(64) void k_fuse(
    const float* __restrict__ y,        // [128][128][128] one batch
    const float* __restrict__ fw,       // [4][3][3][128][16]
    const float* __restrict__ fb,       // [4][16]
    float* __restrict__ out)            // [128][128][64] one batch
{
  const int tid = threadIdx.x;
  const int ri = tid >> 4, oc = tid & 15;
  const int rate = 1 << ri;
  const int i = blockIdx.x >> 7, j = blockIdx.x & 127;
  float acc = fb[ri * 16 + oc];
#pragma unroll
  for (int di = 0; di < 3; ++di) {
    const int yi = i + (di - 1) * rate;
    if (yi < 0 || yi >= H1) continue;
#pragma unroll
    for (int dj = 0; dj < 3; ++dj) {
      const int yj = j + (dj - 1) * rate;
      if (yj < 0 || yj >= H1) continue;
      const float* yp = &y[(size_t)(yi * H1 + yj) * CCH];
      const float* wp = &fw[(size_t)(((ri * 3 + di) * 3 + dj) * CCH) * 16 + oc];
#pragma unroll 8
      for (int c = 0; c < CCH; ++c) acc += yp[c] * wp[c * 16];
    }
  }
  out[(size_t)blockIdx.x * 64 + tid] = fmaxf(acc, 0.f);
}

// ---------------------------------------------------------------------------
extern "C" void kernel_launch(void* const* d_in, const int* in_sizes, int n_in,
                              void* d_out, int out_size, void* d_ws, size_t ws_size,
                              hipStream_t stream) {
  const float* x1   = (const float*)d_in[0];   // [4,128,128,128]
  const float* x2   = (const float*)d_in[1];   // [4,64,64,128]
  const float* mask = (const float*)d_in[2];   // [4,64,64,1]
  const float* fw   = (const float*)d_in[3];   // [4,3,3,128,16]
  const float* fb   = (const float*)d_in[4];   // [4,16]
  float* out = (float*)d_out;                  // [4,128,128,64]

  // Workspace layout (reused across batches; ~147 MB total)
  char* ws = (char*)d_ws;
  size_t off = 0;
  auto carve = [&](size_t bytes) -> char* {
    char* p = ws + off;
    off = (off + bytes + 255) & ~(size_t)255;
    return p;
  };
  unsigned short* A2 = (unsigned short*)carve((size_t)L2PATCH * K9C * 2);      // 9.4 MB
  unsigned short* A1 = (unsigned short*)carve((size_t)L2PATCH * K9C * 2);      // 9.4 MB
  float* inv_norm    = (float*)carve((size_t)L2PATCH * 4);
  float* mi          = (float*)carve((size_t)L2PATCH * 4);
  float* S           = (float*)carve((size_t)L2PATCH * L2PATCH * 4);           // 64 MB
  unsigned short* Sb = (unsigned short*)carve((size_t)L2PATCH * L2PATCH * 2);  // 32 MB
  float* colbuf      = (float*)carve((size_t)L2PATCH * K9C * 4);               // 18.9 MB
  float* ybuf        = (float*)carve((size_t)H1 * H1 * CCH * 4);               // 8.4 MB
  (void)ws_size; (void)in_sizes; (void)n_in; (void)out_size;

  for (int b = 0; b < 4; ++b) {
    k_im2col_x2<<<L2PATCH, 128, 0, stream>>>(x2, b, A2, inv_norm);
    k_im2col_x1<<<L2PATCH, 128, 0, stream>>>(x1, b, A1);
    k_mask_mi<<<16, 256, 0, stream>>>(mask, b, mi);
    k_gemm_scores<<<dim3(32, 32), 256, 0, stream>>>(A2, S);
    k_softmax<<<L2PATCH, 256, 0, stream>>>(S, inv_norm, mi, Sb);
    k_gemm_apply<<<dim3(32, 9), 256, 0, stream>>>(Sb, A1, colbuf);
    k_col2im<<<(H1 * H1 * CCH + 255) / 256, 256, 0, stream>>>(colbuf, ybuf);
    k_fuse<<<H1 * H1, 64, 0, stream>>>(ybuf, fw, fb,
                                       out + (size_t)b * H1 * H1 * 64);
  }
}
